// MultiHeadedSelfAttention_20220706029935
// MI455X (gfx1250) — matrix operand order repaired and checked by reading, not compiled, not run on hardware
//
#include <hip/hip_runtime.h>
#include <hip/hip_bf16.h>
#include <math.h>

typedef __attribute__((ext_vector_type(16))) _Float16 v16h;
typedef __attribute__((ext_vector_type(8)))  _Float16 h8v;
typedef __attribute__((ext_vector_type(8)))  float    v8f;

#define BATCH 16
#define SEQ   1024
#define EMB   512
#define HEADS 8
#define DH    64
#define ROWS  (BATCH*SEQ)          // 16384

__device__ __forceinline__ v8f wmma_f16(v16h a, v16h b, v8f c) {
    return __builtin_amdgcn_wmma_f32_16x16x32_f16(false, a, false, b, (short)0, c, false, false);
}

// A-fragment (16x32 f16, row-major, row stride ld halfs).
// lanes 0-15 (row m=lane): K={0..7,16..23}; lanes 16-31 (row m=lane-16): K={8..15,24..31}.
__device__ __forceinline__ v16h load_a(const _Float16* base, int ld, int lane) {
    int m  = lane & 15;
    int k0 = (lane & 16) ? 8 : 0;
    const _Float16* p = base + (long)m * ld + k0;
    union { v16h v; h8v h[2]; } u;
    u.h[0] = *(const h8v*)(p);
    u.h[1] = *(const h8v*)(p + 16);
    return u.v;
}

// B-fragment (32x16 f16). Column n contiguous in memory (stride cs between columns).
// lanes 0-15: K={0..7,16..23} of column n=lane; lanes 16-31: K={8..15,24..31} of column n=lane-16.
__device__ __forceinline__ v16h load_b(const _Float16* base, long cs, int lane) {
    int n  = lane & 15;
    int k0 = (lane & 16) ? 8 : 0;
    const _Float16* p = base + (long)n * cs + k0;
    union { v16h v; h8v h[2]; } u;
    u.h[0] = *(const h8v*)(p);
    u.h[1] = *(const h8v*)(p + 16);
    return u.v;
}

// gfx1250 async global->LDS copy, 16 bytes per lane. Tracked by ASYNCcnt.
// VDST = per-lane LDS byte address, VADDR = per-lane global address (GV mode).
__device__ __forceinline__ void async_g2l_b128(unsigned lds_byte, const void* gaddr) {
    asm volatile("global_load_async_to_lds_b128 %0, %1, off"
                 :: "v"(lds_byte), "v"(gaddr) : "memory");
}
__device__ __forceinline__ void wait_async0() {
    asm volatile("s_wait_asynccnt 0" ::: "memory");
}

// ---------------------------------------------------------------- cast kernel
__global__ void cvt_f32_f16(const float* __restrict__ in, _Float16* __restrict__ out, int n) {
    int i = blockIdx.x * blockDim.x + threadIdx.x;
    if (i < n) out[i] = (_Float16)in[i];
}

// ---------------------------------------------------------------- QKV projection
// C[16384,1536] = X[16384,512]*W^T. One wave = 32(M) x 64(N) tile, register-double-buffered K loop.
__global__ __launch_bounds__(128) void qkv_gemm(
        const _Float16* __restrict__ X, const _Float16* __restrict__ W,
        const float* __restrict__ bias, const float* __restrict__ qs_scale,
        _Float16* __restrict__ Q, _Float16* __restrict__ K, _Float16* __restrict__ Vt) {
    int lane = threadIdx.x & 31;
    int wv   = blockIdx.x * 4 + (threadIdx.x >> 5);
    int mtile  = wv / 24;          // 512 tiles of 32 rows
    int nstrip = wv % 24;          // 24 strips of 64 over N=1536
    const _Float16* A0 = X + (long)mtile * 32 * EMB;
    const _Float16* A1 = A0 + 16 * EMB;
    const _Float16* WB = W + (long)nstrip * 64 * EMB;

    v8f acc[2][4] = {};
    v16h a0 = load_a(A0, EMB, lane);
    v16h a1 = load_a(A1, EMB, lane);
    v16h b0 = load_b(WB + 0 * 16 * EMB, EMB, lane);
    v16h b1 = load_b(WB + 1 * 16 * EMB, EMB, lane);
    v16h b2 = load_b(WB + 2 * 16 * EMB, EMB, lane);
    v16h b3 = load_b(WB + 3 * 16 * EMB, EMB, lane);

    for (int k0 = 0; k0 < EMB; k0 += 32) {
        int k1 = k0 + 32;
        v16h na0, na1, nb0, nb1, nb2, nb3;
        if (k1 < EMB) {                       // prefetch next k-step
            na0 = load_a(A0 + k1, EMB, lane);
            na1 = load_a(A1 + k1, EMB, lane);
            nb0 = load_b(WB + 0 * 16 * EMB + k1, EMB, lane);
            nb1 = load_b(WB + 1 * 16 * EMB + k1, EMB, lane);
            nb2 = load_b(WB + 2 * 16 * EMB + k1, EMB, lane);
            nb3 = load_b(WB + 3 * 16 * EMB + k1, EMB, lane);
        }
        acc[0][0] = wmma_f16(a0, b0, acc[0][0]);
        acc[1][0] = wmma_f16(a1, b0, acc[1][0]);
        acc[0][1] = wmma_f16(a0, b1, acc[0][1]);
        acc[1][1] = wmma_f16(a1, b1, acc[1][1]);
        acc[0][2] = wmma_f16(a0, b2, acc[0][2]);
        acc[1][2] = wmma_f16(a1, b2, acc[1][2]);
        acc[0][3] = wmma_f16(a0, b3, acc[0][3]);
        acc[1][3] = wmma_f16(a1, b3, acc[1][3]);
        if (k1 < EMB) { a0 = na0; a1 = na1; b0 = nb0; b1 = nb1; b2 = nb2; b3 = nb3; }
    }

    int half = (lane >> 4) & 1;
    int n    = lane & 15;
#pragma unroll
    for (int g = 0; g < 2; ++g) {
        int row0 = mtile * 32 + g * 16;
        int bidx = row0 >> 10;
        int l0   = row0 & 1023;
#pragma unroll
        for (int t = 0; t < 4; ++t) {
            int f   = nstrip * 64 + t * 16 + n;
            float bv = bias[f];
            int sec = f >> 9;      // 0=q 1=k 2=v
            int fe  = f & 511;
            int h   = fe >> 6;
            int dh  = fe & 63;
            long bh = (long)bidx * HEADS + h;
            if (sec == 0) {
                float sc = 0.125f * 1.442695041f * log1pf(__expf(qs_scale[dh]));
#pragma unroll
                for (int r = 0; r < 8; ++r) {
                    int l = l0 + r + half * 8;
                    Q[(bh * SEQ + l) * DH + dh] = (_Float16)((acc[g][t][r] + bv) * sc);
                }
            } else if (sec == 1) {
#pragma unroll
                for (int r = 0; r < 8; ++r) {
                    int l = l0 + r + half * 8;
                    K[(bh * SEQ + l) * DH + dh] = (_Float16)(acc[g][t][r] + bv);
                }
            } else {
                h8v pk;
#pragma unroll
                for (int r = 0; r < 8; ++r) pk[r] = (_Float16)(acc[g][t][r] + bv);
                *(h8v*)(Vt + (bh * DH + dh) * SEQ + l0 + half * 8) = pk;
            }
        }
    }
}

// ---------------------------------------------------------------- flash attention
// Block = 4 waves sharing one (b,h); K/V key-blocks staged in double-buffered LDS
// via async global->LDS copies (ASYNCcnt), shared by all 4 waves.
__global__ __launch_bounds__(128) void attn(
        const _Float16* __restrict__ Q, const _Float16* __restrict__ K,
        const _Float16* __restrict__ Vt, const unsigned char* __restrict__ mask,
        _Float16* __restrict__ Aout) {
    __shared__ __align__(16) _Float16 Ks[2][32 * 64];   // [buf][key][d]
    __shared__ __align__(16) _Float16 Vs[2][64 * 32];   // [buf][dh][key]
    __shared__ __align__(16) _Float16 Plds[4][16 * 32]; // per-wave P tile

    int tid  = threadIdx.x;
    int lane = tid & 31;
    int wid  = tid >> 5;
    int bh   = blockIdx.x >> 4;
    int qblk = blockIdx.x & 15;
    int b    = bh >> 3;
    int h    = bh & 7;
    int l0   = qblk * 64 + wid * 16;

    const _Float16* Qb = Q  + ((long)bh * SEQ + l0) * DH;
    const _Float16* Kb = K  + (long)bh * SEQ * DH;
    const _Float16* Vb = Vt + (long)bh * DH * SEQ;
    const unsigned char* mb = mask + (long)b * SEQ;
    _Float16* pl = &Plds[wid][0];

    // cooperative stage of one 32-key block into LDS buffer `buf`
    auto stage = [&](int buf, int s) {
        // K tile: keys s..s+31 are 4KB contiguous in memory ([l][dh] layout)
        const _Float16* gk = Kb + (long)s * DH;
        unsigned lk = (unsigned)(size_t)(&Ks[buf][0]);
        async_g2l_b128(lk + (unsigned)tid * 16u,         gk + tid * 8);
        async_g2l_b128(lk + (unsigned)(tid + 128) * 16u, gk + (tid + 128) * 8);
        // V tile: column dh = 32 contiguous keys from Vt row (bh*64+dh)
        unsigned lv = (unsigned)(size_t)(&Vs[buf][0]);
        int i0 = tid,        dh0 = i0 >> 2, p0 = i0 & 3;
        int i1 = tid + 128,  dh1 = i1 >> 2, p1 = i1 & 3;
        async_g2l_b128(lv + (unsigned)(dh0 * 64 + p0 * 16), Vb + dh0 * SEQ + s + p0 * 8);
        async_g2l_b128(lv + (unsigned)(dh1 * 64 + p1 * 16), Vb + dh1 * SEQ + s + p1 * 8);
    };

    stage(0, 0);

    v16h qa0 = load_a(Qb + 0,  DH, lane);
    v16h qa1 = load_a(Qb + 32, DH, lane);

    v8f o[4] = {};
    float mrun[8], lrun[8];
#pragma unroll
    for (int r = 0; r < 8; ++r) { mrun[r] = -3.0e38f; lrun[r] = 0.f; }

    int half = (lane >> 4) & 1;
    int n    = lane & 15;

    for (int s = 0, it = 0; s < SEQ; s += 32, ++it) {
        int buf = it & 1;
        wait_async0();          // my async copies for `buf` are done
        __syncthreads();        // everyone's copies done; prev buffer free
        if (s + 32 < SEQ) stage(buf ^ 1, s + 32);

        const _Float16* Kt = &Ks[buf][0];
        const _Float16* Vc = &Vs[buf][0];

        // scores: two 16x16 tiles over keys, Dh reduced in 2 WMMA steps each
        v8f s0 = {}, s1 = {};
        {
            v16h kb0 = load_b(Kt + 0,            64, lane);
            s0 = wmma_f16(qa0, kb0, s0);
            v16h kb1 = load_b(Kt + 32,           64, lane);
            s0 = wmma_f16(qa1, kb1, s0);
            v16h kb2 = load_b(Kt + 16 * 64 + 0,  64, lane);
            s1 = wmma_f16(qa0, kb2, s1);
            v16h kb3 = load_b(Kt + 16 * 64 + 32, 64, lane);
            s1 = wmma_f16(qa1, kb3, s1);
        }
        if (mb[s + n])      { for (int r = 0; r < 8; ++r) s0[r] += -1.0e9f; }
        if (mb[s + 16 + n]) { for (int r = 0; r < 8; ++r) s1[r] += -1.0e9f; }

        // online softmax: row (r, lane-half) spread across 16 lanes
#pragma unroll
        for (int r = 0; r < 8; ++r) {
            float mx = fmaxf(s0[r], s1[r]);
            mx = fmaxf(mx, __shfl_xor(mx, 8, 32));
            mx = fmaxf(mx, __shfl_xor(mx, 4, 32));
            mx = fmaxf(mx, __shfl_xor(mx, 2, 32));
            mx = fmaxf(mx, __shfl_xor(mx, 1, 32));
            float mn    = fmaxf(mrun[r], mx);
            float alpha = __expf(mrun[r] - mn);
            float p0 = __expf(s0[r] - mn);
            float p1 = __expf(s1[r] - mn);
            float rs = p0 + p1;
            rs += __shfl_xor(rs, 8, 32);
            rs += __shfl_xor(rs, 4, 32);
            rs += __shfl_xor(rs, 2, 32);
            rs += __shfl_xor(rs, 1, 32);
            lrun[r] = lrun[r] * alpha + rs;
            mrun[r] = mn;
#pragma unroll
            for (int t = 0; t < 4; ++t) o[t][r] *= alpha;
            int rowb = (r + half * 8) * 32;
            pl[rowb + n]      = (_Float16)p0;
            pl[rowb + 16 + n] = (_Float16)p1;
        }
        asm volatile("s_wait_dscnt 0" ::: "memory");

        // P*V from LDS
        v16h pa = load_a(pl, 32, lane);
#pragma unroll
        for (int t = 0; t < 4; ++t) {
            v16h vb = load_b(Vc + (t * 16) * 32, 32, lane);
            o[t] = wmma_f16(pa, vb, o[t]);
        }
    }

    // normalize + store [B,L,E] f16
#pragma unroll
    for (int t = 0; t < 4; ++t) {
#pragma unroll
        for (int r = 0; r < 8; ++r) {
            int l = l0 + r + half * 8;
            float val = o[t][r] / lrun[r];
            Aout[((long)b * SEQ + l) * EMB + h * DH + t * 16 + n] = (_Float16)val;
        }
    }
}

// ---------------------------------------------------------------- output projection
__global__ __launch_bounds__(128) void out_gemm(
        const _Float16* __restrict__ A, const _Float16* __restrict__ W,
        const float* __restrict__ bias, float* __restrict__ out) {
    int lane = threadIdx.x & 31;
    int wv   = blockIdx.x * 4 + (threadIdx.x >> 5);
    int mtile  = wv >> 3;          // 512 tiles of 32 rows
    int nstrip = wv & 7;           // 8 strips of 64 over N=512
    const _Float16* A0 = A + (long)mtile * 32 * EMB;
    const _Float16* A1 = A0 + 16 * EMB;
    const _Float16* WB = W + (long)nstrip * 64 * EMB;

    v8f acc[2][4] = {};
    v16h a0 = load_a(A0, EMB, lane);
    v16h a1 = load_a(A1, EMB, lane);
    v16h b0 = load_b(WB + 0 * 16 * EMB, EMB, lane);
    v16h b1 = load_b(WB + 1 * 16 * EMB, EMB, lane);
    v16h b2 = load_b(WB + 2 * 16 * EMB, EMB, lane);
    v16h b3 = load_b(WB + 3 * 16 * EMB, EMB, lane);

    for (int k0 = 0; k0 < EMB; k0 += 32) {
        int k1 = k0 + 32;
        v16h na0, na1, nb0, nb1, nb2, nb3;
        if (k1 < EMB) {
            na0 = load_a(A0 + k1, EMB, lane);
            na1 = load_a(A1 + k1, EMB, lane);
            nb0 = load_b(WB + 0 * 16 * EMB + k1, EMB, lane);
            nb1 = load_b(WB + 1 * 16 * EMB + k1, EMB, lane);
            nb2 = load_b(WB + 2 * 16 * EMB + k1, EMB, lane);
            nb3 = load_b(WB + 3 * 16 * EMB + k1, EMB, lane);
        }
        acc[0][0] = wmma_f16(a0, b0, acc[0][0]);
        acc[1][0] = wmma_f16(a1, b0, acc[1][0]);
        acc[0][1] = wmma_f16(a0, b1, acc[0][1]);
        acc[1][1] = wmma_f16(a1, b1, acc[1][1]);
        acc[0][2] = wmma_f16(a0, b2, acc[0][2]);
        acc[1][2] = wmma_f16(a1, b2, acc[1][2]);
        acc[0][3] = wmma_f16(a0, b3, acc[0][3]);
        acc[1][3] = wmma_f16(a1, b3, acc[1][3]);
        if (k1 < EMB) { a0 = na0; a1 = na1; b0 = nb0; b1 = nb1; b2 = nb2; b3 = nb3; }
    }

    int half = (lane >> 4) & 1;
    int n    = lane & 15;
#pragma unroll
    for (int g = 0; g < 2; ++g) {
        int row0 = mtile * 32 + g * 16;
#pragma unroll
        for (int t = 0; t < 4; ++t) {
            int f = nstrip * 64 + t * 16 + n;
            float bv = bias[f];
#pragma unroll
            for (int r = 0; r < 8; ++r) {
                out[((long)row0 + r + half * 8) * EMB + f] = acc[g][t][r] + bv;
            }
        }
    }
}

// ---------------------------------------------------------------- launcher
extern "C" void kernel_launch(void* const* d_in, const int* in_sizes, int n_in,
                              void* d_out, int out_size, void* d_ws, size_t ws_size,
                              hipStream_t stream) {
    (void)in_sizes; (void)n_in; (void)out_size; (void)ws_size;
    const float* x   = (const float*)d_in[0];
    const float* ipw = (const float*)d_in[1];
    const float* ipb = (const float*)d_in[2];
    const float* opw = (const float*)d_in[3];
    const float* opb = (const float*)d_in[4];
    const float* qsc = (const float*)d_in[5];
    const unsigned char* kpm = (const unsigned char*)d_in[6];
    float* out = (float*)d_out;

    char* ws = (char*)d_ws;
    const long N_X  = (long)ROWS * EMB;        // 8,388,608
    const long N_W  = 3L * EMB * EMB;          // 786,432
    const long N_OW = (long)EMB * EMB;         // 262,144
    _Float16* X16  = (_Float16*)(ws);
    _Float16* W16  = (_Float16*)(ws + 2 * N_X);
    _Float16* OW16 = (_Float16*)(ws + 2 * (N_X + N_W));
    _Float16* Q16  = (_Float16*)(ws + 2 * (N_X + N_W + N_OW));
    _Float16* K16  = Q16  + N_X;
    _Float16* V16t = K16  + N_X;
    _Float16* A16  = V16t + N_X;

    cvt_f32_f16<<<(int)((N_X  + 255) / 256), 256, 0, stream>>>(x,   X16,  (int)N_X);
    cvt_f32_f16<<<(int)((N_W  + 255) / 256), 256, 0, stream>>>(ipw, W16,  (int)N_W);
    cvt_f32_f16<<<(int)((N_OW + 255) / 256), 256, 0, stream>>>(opw, OW16, (int)N_OW);

    // QKV: 512 m-tiles(32 rows) * 24 n-strips = 12288 waves / 4 per block
    qkv_gemm<<<3072, 128, 0, stream>>>(X16, W16, ipb, qsc, Q16, K16, V16t);

    // attention: 128 (b,h) * 16 query blocks of 64 rows
    attn<<<2048, 128, 0, stream>>>(Q16, K16, V16t, kpm, A16);

    // out proj: 512 m-tiles * 8 n-strips = 4096 waves / 4 per block
    out_gemm<<<1024, 128, 0, stream>>>(A16, OW16, opb, out);
}
